// UtterNetwork_3367254360564
// MI455X (gfx1250) — compile-verified, matrix operand order checked by reading, not run on hardware
//
#include <hip/hip_runtime.h>
#include <hip/hip_bf16.h>
#include <hip/hip_fp16.h>

// ---------------------------------------------------------------------------
// CDNA5 (gfx1250) WMMA helpers: 16x16x32 f16 -> f32, wave32.
// A/B fragment layout (ISA 7.12.2, 16-bit A 16x32): lane L<16 holds row
// rowBase+L, K = {k0..k0+7, k0+16..k0+23}; lane L>=16 holds same row with
// K shifted by +8. B (K-major per column) uses the identical pattern over
// rows of the pre-transposed weight (Bt[n][k] = B[k][n]).
// C/D layout: VGPR j -> (row rowBase + j + 8*(lane>=16), col colBase+(lane&15)).
// ---------------------------------------------------------------------------
typedef __attribute__((ext_vector_type(16))) _Float16 v16h;
typedef __attribute__((ext_vector_type(8)))  float    v8f;

union FragH { v16h v; _Float16 h[16]; };
union FragF { v8f  v; float     f[8]; };

__device__ inline v16h wmma_load_frag(const _Float16* p, int ld, int rowBase, int k0) {
  int lane = threadIdx.x & 31;
  const _Float16* s = p + (size_t)(rowBase + (lane & 15)) * ld + k0 + ((lane >> 4) << 3);
  FragH f;
#pragma unroll
  for (int i = 0; i < 8; ++i) { f.h[i] = s[i]; f.h[8 + i] = s[16 + i]; }
  return f.v;
}

__device__ inline void wmma_store_tile(float* C, int ldc, int rowBase, int colBase, FragF acc) {
  int lane = threadIdx.x & 31;
  int r0 = rowBase + ((lane >> 4) << 3);
  int c  = colBase + (lane & 15);
#pragma unroll
  for (int j = 0; j < 8; ++j) C[(size_t)(r0 + j) * ldc + c] = acc.f[j];
}

// Generic-pointer -> 0-based LDS byte offset (for async-to-LDS VDST).
__device__ inline unsigned lds_offset(const void* p) {
  return (unsigned)(unsigned long long)(__attribute__((address_space(3))) const char*)p;
}

// ---------------------------------------------------------------------------
// LDS-tiled GEMM: C(M,N) = A(M,K) * Bt(N,K)^T.
// Grid: (N/16, M/128); block = 256 threads = 8 waves.
// The 16xK weight panel for this block's N-tile is staged into LDS ONCE via
// CDNA5 async copy (global_load_async_to_lds_b128, ASYNCcnt), then all 8
// waves (one 16x16 output tile each, stacked in M) read B-fragments from LDS
// while streaming their private A rows from global. K%32==0, K<=320.
// ---------------------------------------------------------------------------
__global__ void __launch_bounds__(256)
gemm_f16_wmma_lds(const _Float16* __restrict__ A, int lda,
                  const _Float16* __restrict__ Bt, int ldb,
                  float* __restrict__ C, int ldc,
                  int M, int N, int K) {
  __shared__ __align__(16) _Float16 Bs[16 * 320];
  (void)M; (void)N;

  int tn = blockIdx.x;                 // N/16 tile index
  int rowPanel = blockIdx.y << 7;      // 128-row M panel

  // ---- stage Bt[tn*16 .. tn*16+15][0..K) into LDS via async copy ----
  int chunksPerRow = K >> 3;           // 16-byte chunks per row
  int chunks = 16 * chunksPerRow;
  unsigned ldsBase = lds_offset(Bs);
  for (int i = threadIdx.x; i < chunks; i += blockDim.x) {
    int r = i / chunksPerRow;
    int c = (i - r * chunksPerRow) << 3;
    unsigned long long g = (unsigned long long)(size_t)(Bt + (size_t)((tn << 4) + r) * ldb + c);
    unsigned lds = ldsBase + (unsigned)((r * K + c) << 1);
    asm volatile("global_load_async_to_lds_b128 %0, %1, off"
                 :: "v"(lds), "v"(g) : "memory");
  }
  asm volatile("s_wait_asynccnt 0x0" ::: "memory");
  __syncthreads();

  // ---- each wave: one 16x16 tile, B from LDS, A from global ----
  int wave = threadIdx.x >> 5;
  int rowBase = rowPanel + (wave << 4);
  FragF acc;
#pragma unroll
  for (int j = 0; j < 8; ++j) acc.f[j] = 0.0f;
  for (int k0 = 0; k0 < K; k0 += 32) {
    v16h a = wmma_load_frag(A,  lda, rowBase, k0);
    v16h b = wmma_load_frag(Bs, K,   0,       k0);   // ds_load path
    acc.v = __builtin_amdgcn_wmma_f32_16x16x32_f16(false, a, false, b,
                                                   (short)0, acc.v, false, false);
  }
  wmma_store_tile(C, ldc, rowBase, tn << 4, acc);
}

// ---------------------------------------------------------------------------
// Embedding gather-sum: X0[row][0:300] = sum_w table[ids[row][w]], pad to 320.
// ---------------------------------------------------------------------------
__global__ void embed_sum_f16(const int* __restrict__ ids,
                              const float* __restrict__ table,
                              _Float16* __restrict__ X, int NW, int ED, int EDP) {
  __shared__ int sid[64];
  int row = blockIdx.x;
  if (threadIdx.x < NW) sid[threadIdx.x] = ids[(size_t)row * NW + threadIdx.x];
  __syncthreads();
  for (int d = threadIdx.x; d < EDP; d += blockDim.x) {
    float s = 0.0f;
    if (d < ED) {
      for (int w = 0; w < NW; ++w) s += table[(size_t)sid[w] * ED + d];
    }
    X[(size_t)row * EDP + d] = (_Float16)s;
  }
}

// ---------------------------------------------------------------------------
// Conversions / small utilities
// ---------------------------------------------------------------------------
__global__ void f32_to_f16_pad(const float* __restrict__ src, _Float16* __restrict__ dst,
                               int rows, int cs, int cd) {
  int i = blockIdx.x * blockDim.x + threadIdx.x;
  if (i >= rows * cd) return;
  int r = i / cd, c = i % cd;
  dst[i] = (_Float16)(c < cs ? src[(size_t)r * cs + c] : 0.0f);
}

__global__ void f32_to_f16_transpose(const float* __restrict__ src, _Float16* __restrict__ dst,
                                     int rows, int cols) {
  int i = blockIdx.x * blockDim.x + threadIdx.x;
  if (i >= rows * cols) return;
  int r = i / cols, c = i % cols;
  dst[(size_t)c * rows + r] = (_Float16)src[i];
}

__global__ void rowmask_f32_to_f16(const float* __restrict__ src,
                                   const float* __restrict__ rowMask,
                                   _Float16* __restrict__ dst, int n, int C) {
  int i = blockIdx.x * blockDim.x + threadIdx.x;
  if (i >= n) return;
  float m = rowMask ? rowMask[i / C] : 1.0f;
  dst[i] = (_Float16)(src[i] * m);
}

__global__ void fill_f32(float* __restrict__ p, float v, int n) {
  int i = blockIdx.x * blockDim.x + threadIdx.x;
  if (i < n) p[i] = v;
}

// ---------------------------------------------------------------------------
// BiGRU recurrence (one direction per workgroup, blockIdx.x = 0:fwd, 1:bwd).
// Hidden state lives in LDS; per-step gh = h @ Whh^T via WMMA (2x24 tiles, K=128).
// gi (= X@Wih^T for all timesteps) precomputed in global memory.
// ---------------------------------------------------------------------------
__global__ void __launch_bounds__(256)
gru_recurrence(const float* __restrict__ giF, const float* __restrict__ giB,
               const _Float16* __restrict__ WhhF, const _Float16* __restrict__ WhhB,
               const float* __restrict__ bihF, const float* __restrict__ bihB,
               const float* __restrict__ bhhF, const float* __restrict__ bhhB,
               float* __restrict__ out) {            // out: (2048, 256), fwd cols 0:128, bwd 128:256
  const int BB = 32, UU = 64, HH = 128, GG = 384;
  __shared__ __align__(16) _Float16 hF16[BB * HH];   // 8 KB
  __shared__ float    hF32[BB * HH];                 // 16 KB
  __shared__ float    ghS [BB * GG];                 // 48 KB

  int dir = blockIdx.x;
  const float*    gi  = dir ? giB  : giF;
  const _Float16* Whh = dir ? WhhB : WhhF;
  const float*    bih = dir ? bihB : bihF;
  const float*    bhh = dir ? bhhB : bhhF;
  int outOff = dir * HH;

  for (int e = threadIdx.x; e < BB * HH; e += blockDim.x) {
    hF16[e] = (_Float16)0.0f;
    hF32[e] = 0.0f;
  }
  __syncthreads();

  int waveId = threadIdx.x >> 5;
  for (int tt = 0; tt < UU; ++tt) {
    int t = dir ? (UU - 1 - tt) : tt;
    // ---- phase 1: gh = h @ Whh^T  (M=32, N=384, K=128) ----
    for (int tile = waveId; tile < 2 * (GG >> 4); tile += 8) {   // 48 tiles / 8 waves
      int tm = tile / (GG >> 4), tn = tile % (GG >> 4);
      FragF acc;
#pragma unroll
      for (int j = 0; j < 8; ++j) acc.f[j] = 0.0f;
#pragma unroll
      for (int k0 = 0; k0 < HH; k0 += 32) {
        v16h a = wmma_load_frag(hF16, HH, tm << 4, k0);
        v16h b = wmma_load_frag(Whh,  HH, tn << 4, k0);
        acc.v = __builtin_amdgcn_wmma_f32_16x16x32_f16(false, a, false, b,
                                                       (short)0, acc.v, false, false);
      }
      wmma_store_tile(ghS, GG, tm << 4, tn << 4, acc);
    }
    __syncthreads();
    // ---- phase 2: gates + state update ----
    for (int e = threadIdx.x; e < BB * HH; e += blockDim.x) {
      int b = e >> 7, j = e & 127;
      size_t gbase = (size_t)(b * UU + t) * GG;
      float ir = gi[gbase + j]           + bih[j];
      float iz = gi[gbase + HH + j]      + bih[HH + j];
      float in = gi[gbase + 2 * HH + j]  + bih[2 * HH + j];
      float hr = ghS[b * GG + j]          + bhh[j];
      float hz = ghS[b * GG + HH + j]     + bhh[HH + j];
      float hn = ghS[b * GG + 2 * HH + j] + bhh[2 * HH + j];
      float r = 1.0f / (1.0f + __expf(-(ir + hr)));
      float z = 1.0f / (1.0f + __expf(-(iz + hz)));
      float n = tanhf(in + r * hn);
      float hnew = (1.0f - z) * n + z * hF32[e];
      hF32[e] = hnew;
      hF16[e] = (_Float16)hnew;
      out[(size_t)(b * UU + t) * 256 + outOff + j] = hnew;
    }
    __syncthreads();
  }
}

// ---------------------------------------------------------------------------
// GCN graph pieces
// ---------------------------------------------------------------------------
__global__ void edge_deg(const int* __restrict__ dst, float* __restrict__ deg, int E) {
  int e = blockIdx.x * blockDim.x + threadIdx.x;
  if (e < E) atomicAdd(&deg[dst[e]], 1.0f);
}

__global__ void deg_to_dis(float* __restrict__ deg, int n) {
  int i = blockIdx.x * blockDim.x + threadIdx.x;
  if (i < n) { float d = deg[i]; deg[i] = d > 0.0f ? rsqrtf(d) : 0.0f; }
}

__global__ void gcn_scatter(const float* __restrict__ xW,
                            const int* __restrict__ src, const int* __restrict__ dst,
                            const float* __restrict__ dis,
                            float* __restrict__ agg, int C) {
  int e = blockIdx.x;
  int s = src[e], d = dst[e];
  float w = dis[s] * dis[d];
  for (int j = threadIdx.x; j < C; j += blockDim.x)
    atomicAdd(&agg[(size_t)d * C + j], xW[(size_t)s * C + j] * w);
}

__global__ void bias_act(const float* __restrict__ agg, const float* __restrict__ bias,
                         _Float16* __restrict__ outH, float* __restrict__ outF,
                         int relu, int n, int C) {
  int i = blockIdx.x * blockDim.x + threadIdx.x;
  if (i >= n) return;
  float v = agg[i] + bias[i % C];
  if (relu) v = fmaxf(v, 0.0f);
  if (outH) outH[i] = (_Float16)v;
  if (outF) outF[i] = v;
}

__global__ void batch_sum(const float* __restrict__ gx, const int* __restrict__ bvec,
                          float* __restrict__ out2, int C) {
  int row = blockIdx.x;
  int j = threadIdx.x;
  atomicAdd(&out2[(size_t)bvec[row] * C + j], gx[(size_t)row * C + j]);
}

// ---------------------------------------------------------------------------
// Orchestration
// ---------------------------------------------------------------------------
extern "C" void kernel_launch(void* const* d_in, const int* in_sizes, int n_in,
                              void* d_out, int out_size, void* d_ws, size_t ws_size,
                              hipStream_t stream) {
  (void)n_in; (void)out_size; (void)ws_size;
  const int NN = 2048, UU = 64, NW = 50, ED = 300, EDP = 320, DD = 256, HH = 128,
            GG = 384, HC = 256, BB = 32;
  (void)UU;

  const int*   ids    = (const int*)  d_in[0];
  const float* umasks = (const float*)d_in[1];
  const int*   edges  = (const int*)  d_in[3];
  const int*   bvec   = (const int*)  d_in[4];
  const float* table  = (const float*)d_in[5];
  const float *Wih0f=(const float*)d_in[6],  *Whh0f=(const float*)d_in[7],
              *bih0f=(const float*)d_in[8],  *bhh0f=(const float*)d_in[9];
  const float *Wih0b=(const float*)d_in[10], *Whh0b=(const float*)d_in[11],
              *bih0b=(const float*)d_in[12], *bhh0b=(const float*)d_in[13];
  const float *Wih1f=(const float*)d_in[14], *Whh1f=(const float*)d_in[15],
              *bih1f=(const float*)d_in[16], *bhh1f=(const float*)d_in[17];
  const float *Wih1b=(const float*)d_in[18], *Whh1b=(const float*)d_in[19],
              *bih1b=(const float*)d_in[20], *bhh1b=(const float*)d_in[21];
  const float *W1=(const float*)d_in[22], *b1=(const float*)d_in[23];
  const float *W2=(const float*)d_in[24], *b2=(const float*)d_in[25];
  const float *W3=(const float*)d_in[26], *b3=(const float*)d_in[27];
  int E = in_sizes[3] / 2;
  const int* esrc = edges;
  const int* edst = edges + E;

  // bump allocator over d_ws
  char* base = (char*)d_ws; size_t off = 0;
  auto alloc = [&](size_t bytes) -> void* {
    void* p = base + off;
    off += (bytes + 255) & ~(size_t)255;
    return p;
  };

  _Float16* X0     = (_Float16*)alloc((size_t)NN * EDP * 2);
  _Float16* W0fH   = (_Float16*)alloc((size_t)GG * EDP * 2);
  _Float16* W0bH   = (_Float16*)alloc((size_t)GG * EDP * 2);
  _Float16* Wh0fH  = (_Float16*)alloc((size_t)GG * HH * 2);
  _Float16* Wh0bH  = (_Float16*)alloc((size_t)GG * HH * 2);
  _Float16* Wh1fH  = (_Float16*)alloc((size_t)GG * HH * 2);
  _Float16* Wh1bH  = (_Float16*)alloc((size_t)GG * HH * 2);
  _Float16* W1fH   = (_Float16*)alloc((size_t)GG * DD * 2);
  _Float16* W1bH   = (_Float16*)alloc((size_t)GG * DD * 2);
  _Float16* Wt1    = (_Float16*)alloc((size_t)HC * HC * 2);
  _Float16* Wt2    = (_Float16*)alloc((size_t)HC * HC * 2);
  _Float16* Wt3    = (_Float16*)alloc((size_t)HC * HC * 2);
  float*    giF    = (float*)   alloc((size_t)NN * GG * 4);
  float*    giB    = (float*)   alloc((size_t)NN * GG * 4);
  float*    h0     = (float*)   alloc((size_t)NN * DD * 4);
  _Float16* h0H    = (_Float16*)alloc((size_t)NN * DD * 2);
  float*    utter  = (float*)   alloc((size_t)NN * DD * 4);
  _Float16* hA     = (_Float16*)alloc((size_t)NN * HC * 2);
  float*    xW     = (float*)   alloc((size_t)NN * HC * 4);
  float*    agg    = (float*)   alloc((size_t)NN * HC * 4);
  float*    dis    = (float*)   alloc((size_t)NN * 4);

  float* gxOut  = (float*)d_out;               // (2048, 256)
  float* sumOut = gxOut + (size_t)NN * HC;     // (32, 256)

  auto cdiv = [](int a, int b) { return (a + b - 1) / b; };
  auto gemm = [&](const _Float16* A, int lda, const _Float16* Bt, int ldb,
                  float* C, int ldc, int M, int Nc, int K) {
    gemm_f16_wmma_lds<<<dim3(Nc / 16, M / 128), 256, 0, stream>>>(A, lda, Bt, ldb,
                                                                  C, ldc, M, Nc, K);
  };

  // 1) embedding gather-sum -> f16, padded K=320
  embed_sum_f16<<<NN, 128, 0, stream>>>(ids, table, X0, NW, ED, EDP);

  // 2) weight conversions
  f32_to_f16_pad<<<cdiv(GG * EDP, 256), 256, 0, stream>>>(Wih0f, W0fH, GG, ED, EDP);
  f32_to_f16_pad<<<cdiv(GG * EDP, 256), 256, 0, stream>>>(Wih0b, W0bH, GG, ED, EDP);
  f32_to_f16_pad<<<cdiv(GG * HH, 256), 256, 0, stream>>>(Whh0f, Wh0fH, GG, HH, HH);
  f32_to_f16_pad<<<cdiv(GG * HH, 256), 256, 0, stream>>>(Whh0b, Wh0bH, GG, HH, HH);
  f32_to_f16_pad<<<cdiv(GG * HH, 256), 256, 0, stream>>>(Whh1f, Wh1fH, GG, HH, HH);
  f32_to_f16_pad<<<cdiv(GG * HH, 256), 256, 0, stream>>>(Whh1b, Wh1bH, GG, HH, HH);
  f32_to_f16_pad<<<cdiv(GG * DD, 256), 256, 0, stream>>>(Wih1f, W1fH, GG, DD, DD);
  f32_to_f16_pad<<<cdiv(GG * DD, 256), 256, 0, stream>>>(Wih1b, W1bH, GG, DD, DD);
  f32_to_f16_transpose<<<cdiv(HC * HC, 256), 256, 0, stream>>>(W1, Wt1, HC, HC);
  f32_to_f16_transpose<<<cdiv(HC * HC, 256), 256, 0, stream>>>(W2, Wt2, HC, HC);
  f32_to_f16_transpose<<<cdiv(HC * HC, 256), 256, 0, stream>>>(W3, Wt3, HC, HC);

  // 3) GRU layer 0: input projections (async-staged WMMA GEMMs), then LDS recurrence
  gemm(X0, EDP, W0fH, EDP, giF, GG, NN, GG, EDP);
  gemm(X0, EDP, W0bH, EDP, giB, GG, NN, GG, EDP);
  gru_recurrence<<<2, 256, 0, stream>>>(giF, giB, Wh0fH, Wh0bH,
                                        bih0f, bih0b, bhh0f, bhh0b, h0);

  // 4) GRU layer 1
  rowmask_f32_to_f16<<<cdiv(NN * DD, 256), 256, 0, stream>>>(h0, nullptr, h0H, NN * DD, DD);
  gemm(h0H, DD, W1fH, DD, giF, GG, NN, GG, DD);
  gemm(h0H, DD, W1bH, DD, giB, GG, NN, GG, DD);
  gru_recurrence<<<2, 256, 0, stream>>>(giF, giB, Wh1fH, Wh1bH,
                                        bih1f, bih1b, bhh1f, bhh1b, utter);

  // 5) mask + f16 for GCN input
  rowmask_f32_to_f16<<<cdiv(NN * DD, 256), 256, 0, stream>>>(utter, umasks, hA, NN * DD, DD);

  // 6) degree -> rsqrt normalization
  fill_f32<<<cdiv(NN, 256), 256, 0, stream>>>(dis, 0.0f, NN);
  edge_deg<<<cdiv(E, 256), 256, 0, stream>>>(edst, dis, E);
  deg_to_dis<<<cdiv(NN, 256), 256, 0, stream>>>(dis, NN);

  // 7) GCN layer 1 (relu)
  gemm(hA, HC, Wt1, HC, xW, HC, NN, HC, HC);
  fill_f32<<<cdiv(NN * HC, 256), 256, 0, stream>>>(agg, 0.0f, NN * HC);
  gcn_scatter<<<E, 256, 0, stream>>>(xW, esrc, edst, dis, agg, HC);
  bias_act<<<cdiv(NN * HC, 256), 256, 0, stream>>>(agg, b1, hA, nullptr, 1, NN * HC, HC);

  // 8) GCN layer 2 (relu)
  gemm(hA, HC, Wt2, HC, xW, HC, NN, HC, HC);
  fill_f32<<<cdiv(NN * HC, 256), 256, 0, stream>>>(agg, 0.0f, NN * HC);
  gcn_scatter<<<E, 256, 0, stream>>>(xW, esrc, edst, dis, agg, HC);
  bias_act<<<cdiv(NN * HC, 256), 256, 0, stream>>>(agg, b2, hA, nullptr, 1, NN * HC, HC);

  // 9) GCN layer 3 (no relu) -> gx into d_out
  gemm(hA, HC, Wt3, HC, xW, HC, NN, HC, HC);
  fill_f32<<<cdiv(NN * HC, 256), 256, 0, stream>>>(agg, 0.0f, NN * HC);
  gcn_scatter<<<E, 256, 0, stream>>>(xW, esrc, edst, dis, agg, HC);
  bias_act<<<cdiv(NN * HC, 256), 256, 0, stream>>>(agg, b3, nullptr, gxOut, 0, NN * HC, HC);

  // 10) per-batch segment sum -> second output
  fill_f32<<<cdiv(BB * HC, 256), 256, 0, stream>>>(sumOut, 0.0f, BB * HC);
  batch_sum<<<NN, 256, 0, stream>>>(gxOut, bvec, sumOut, HC);
}
